// SpatialMemoryGraphODE_91122026152812
// MI455X (gfx1250) — matrix operand order repaired and checked by reading, not compiled, not run on hardware
//
#include <hip/hip_runtime.h>
#include <hip/hip_bf16.h>
#include <math.h>
#include <stdint.h>

typedef __attribute__((ext_vector_type(16))) _Float16 v16h;
typedef __attribute__((ext_vector_type(8)))  _Float16 v8h;
typedef __attribute__((ext_vector_type(8)))  float    v8f;
typedef __attribute__((ext_vector_type(4)))  unsigned int u32x4;
typedef __attribute__((ext_vector_type(4)))  int          i32x4;
typedef __attribute__((ext_vector_type(8)))  int          i32x8;

// problem dims
static constexpr int DB = 4;        // batch
static constexpr int DN = 2048;     // nodes
static constexpr int DF = 256;      // features
static constexpr int DH = 512;      // hidden
static constexpr int DE = 65536;    // edges
static constexpr int DR = DB * DN;  // 8192 fused rows

// GEMM tiling
static constexpr int GBM = 128;
static constexpr int GBN = 128;
static constexpr int GBK = 32;
static constexpr int LP  = GBK + 8; // padded LDS pitch (f16): 64B row + 16B pad, 16B aligned

// ---------------------------------------------------------------- async copy helpers
__device__ __forceinline__ void async_b128(void* lds_ptr, const void* gptr) {
    uint32_t la = (uint32_t)(uintptr_t)lds_ptr;          // low 32 bits = LDS byte address
    uint64_t ga = (uint64_t)(uintptr_t)gptr;
    asm volatile("global_load_async_to_lds_b128 %0, %1, off"
                 :: "v"(la), "v"(ga) : "memory");
}
__device__ __forceinline__ void wait_async0() { asm volatile("s_wait_asynccnt 0x0" ::: "memory"); }
__device__ __forceinline__ void wait_async2() { asm volatile("s_wait_asynccnt 0x2" ::: "memory"); }

// ---------------------------------------------------------------- TDM tile load
// 2D tile load: GBM rows x GBK f16 from a row-major tensor with row pitch K (elements)
// into LDS with hardware row padding: 64B interval (pad_interval=3 -> 16 DWORDs),
// 16B pad (pad_amount=3 -> 4 DWORDs) => LDS pitch LP=40 f16, matching fragment indexing.
__device__ __forceinline__ void tdm_load_tile(const _Float16* gtile, void* lds, int K) {
    uint64_t ga = (uint64_t)(uintptr_t)gtile;
    uint32_t la = (uint32_t)(uintptr_t)lds;
    u32x4 g0;
    g0[0] = 1u;                                            // count=1 (valid), user mode
    g0[1] = la;                                            // lds_addr
    g0[2] = (uint32_t)ga;                                  // global_addr[31:0]
    g0[3] = (uint32_t)((ga >> 32) & 0x01FFFFFFu) | (2u << 30); // addr[56:32] | type=2
    i32x8 g1;
    g1[0] = (int)((1u << 16) | (1u << 20) | (3u << 22) | (3u << 25)); // data_size=2B, pad_en, ivl=3, amt=3
    g1[1] = (int)((unsigned)GBK << 16);                    // tensor_dim0 = 32
    g1[2] = (int)((unsigned)GBM << 16);                    // tensor_dim1 = 128
    g1[3] = (int)((unsigned)GBK << 16);                    // tile_dim0 = 32
    g1[4] = GBM;                                           // tile_dim1 = 128, tile_dim2 = 0
    g1[5] = K;                                             // tensor_dim0_stride (elements)
    g1[6] = 0;                                             // stride hi / dim1_stride lo
    g1[7] = 0;
    i32x4 z4 = {0, 0, 0, 0};
    i32x8 z8 = {0, 0, 0, 0, 0, 0, 0, 0};
    __builtin_amdgcn_tensor_load_to_lds(g0, g1, z4, z4, z8, 0);   // 6-arg clang-23 form
}

// ---------------------------------------------------------------- utilities
__global__ void fill_f32_kernel(float* p, float v, long n) {
    long i = (long)blockIdx.x * blockDim.x + threadIdx.x;
    if (i < n) p[i] = v;
}
__global__ void copy_f32_kernel(const float* __restrict__ s, float* __restrict__ d, long n) {
    long i = (long)blockIdx.x * blockDim.x + threadIdx.x;
    if (i < n) d[i] = s[i];
}
__global__ void cvt_f32_f16_kernel(const float* __restrict__ s, _Float16* __restrict__ d, long n) {
    long i = (long)blockIdx.x * blockDim.x + threadIdx.x;
    if (i < n) d[i] = (_Float16)s[i];
}

// tiled transpose + f32->f16 cast: src f32 [R][C] -> dst f16 [C][R]; R,C multiples of 32
__global__ __launch_bounds__(256)
void transpose_cvt_kernel(const float* __restrict__ s, _Float16* __restrict__ d,
                          int R, int C, long sSz, long sDz) {
    __shared__ _Float16 t[32][33];
    s += (long)blockIdx.z * sSz;
    d += (long)blockIdx.z * sDz;
    int c0 = blockIdx.x * 32, r0 = blockIdx.y * 32;
    int tr = threadIdx.x >> 5, tc = threadIdx.x & 31;
    #pragma unroll
    for (int p = 0; p < 4; ++p) {
        int r = r0 + tr + p * 8;
        t[tr + p * 8][tc] = (_Float16)s[(long)r * C + c0 + tc];
    }
    __syncthreads();
    #pragma unroll
    for (int p = 0; p < 4; ++p) {
        int cc = c0 + tr + p * 8;
        d[(long)cc * R + r0 + tc] = t[tc][tr + p * 8];
    }
}

// build X = concat(y, broadcast ctx) in f16  [DR, 2F]
__global__ void build_x_kernel(const float* __restrict__ y, const float* __restrict__ ctx,
                               _Float16* __restrict__ X) {
    long i = (long)blockIdx.x * blockDim.x + threadIdx.x;
    long n = (long)DR * (2 * DF);
    if (i >= n) return;
    int col = (int)(i % (2 * DF));
    long row = i / (2 * DF);
    int b = (int)(row / DN);
    float v = (col < DF) ? y[row * DF + col] : ctx[(long)b * DF + (col - DF)];
    X[i] = (_Float16)v;
}

// ---------------------------------------------------------------- graph build
__global__ void edge_mlp_kernel(const float* __restrict__ ea,
                                const float* __restrict__ w1, const float* __restrict__ b1,
                                const float* __restrict__ w2, const float* __restrict__ b2,
                                float* __restrict__ ew) {
    int e = blockIdx.x * blockDim.x + threadIdx.x;
    if (e >= DE) return;
    float a = ea[e];
    float s = b2[0];
    #pragma unroll
    for (int j = 0; j < 32; ++j) {
        float h = fmaxf(a * w1[j] + b1[j], 0.0f);
        s += h * w2[j];
    }
    ew[e] = 1.0f / (1.0f + expf(-s));
}
__global__ void deg_acc_kernel(const int* __restrict__ ei, float* __restrict__ deg) {
    int e = blockIdx.x * blockDim.x + threadIdx.x;
    if (e >= DE) return;
    atomicAdd(&deg[ei[DE + e]], 1.0f);
}
__global__ void dis_diag_kernel(const float* __restrict__ deg, float* __restrict__ dis,
                                float* __restrict__ adjf) {
    int n = blockIdx.x * blockDim.x + threadIdx.x;
    if (n >= DN) return;
    float d = deg[n];
    float di = (d > 0.0f) ? rsqrtf(d) : 0.0f;
    dis[n] = di;
    adjf[(long)n * DN + n] = di * di;   // self loop weight 1
}
__global__ void adj_acc_kernel(const int* __restrict__ ei, const float* __restrict__ ew,
                               const float* __restrict__ dis, float* __restrict__ adjf) {
    int e = blockIdx.x * blockDim.x + threadIdx.x;
    if (e >= DE) return;
    int r = ei[e], c = ei[DE + e];
    float w = dis[r] * ew[e] * dis[c];
    atomicAdd(&adjf[(long)r * DN + c], w);
}

// ---------------------------------------------------------------- WMMA GEMM
// C[M,N] = A[M,K](f16,row-major) * Bt[N,K](f16, pre-transposed) + bias, optional ReLU.
// A tile staged by the Tensor Data Mover (one TENSOR_LOAD_TO_LDS per tile, wave 0,
// TENSORcnt), B tile staged with GLOBAL_LOAD_ASYNC_TO_LDS_B128 (ASYNCcnt); both
// double-buffered so the DMA of tile i+1 overlaps WMMA on tile i.
template <bool RELU>
__global__ __launch_bounds__(256)
void wmma_gemm_kernel(const _Float16* __restrict__ A, const _Float16* __restrict__ Bt,
                      const float* __restrict__ bias, float* __restrict__ C,
                      int M, int Nn, int K,
                      long strideAz, long strideBz, long strideCz) {
    __shared__ _Float16 sA[2][GBM * LP];
    __shared__ _Float16 sB[2][GBN * LP];   // [n][k]

    A  += (long)blockIdx.z * strideAz;
    Bt += (long)blockIdx.z * strideBz;
    C  += (long)blockIdx.z * strideCz;

    const int tid  = threadIdx.x;
    const int lane = tid & 31;
    const int wave = tid >> 5;          // 0..7
    const int waveM = wave & 3;         // 4 along M
    const int waveN = wave >> 2;        // 2 along N
    const int rowBase = blockIdx.y * GBM;
    const int colBase = blockIdx.x * GBN;
    const int wRow = waveM * 32;
    const int wCol = waveN * 64;
    const int halfSel = lane >> 4;      // 0 | 1
    const int mLane   = lane & 15;
    const bool tdmWave = (wave == 0);   // TDM ignores EXEC: issue from one wave only

    v8f acc[2][4];
    #pragma unroll
    for (int mi = 0; mi < 2; ++mi)
        #pragma unroll
        for (int ni = 0; ni < 4; ++ni)
            #pragma unroll
            for (int r = 0; r < 8; ++r) acc[mi][ni][r] = 0.0f;

    // B tile: 512 x 16B chunks, 2 per thread => 2 async instructions per wave per tile
    auto stageB = [&](int buf, int k0) {
        #pragma unroll
        for (int it = 0; it < 2; ++it) {
            int idx = tid + it * 256;
            int r  = idx >> 2;
            int c8 = (idx & 3) * 8;
            async_b128(&sB[buf][r * LP + c8], Bt + (long)(colBase + r) * K + k0 + c8);
        }
    };

    if (tdmWave) tdm_load_tile(A + (long)rowBase * K, &sA[0][0], K);
    stageB(0, 0);

    for (int k0 = 0, itn = 0; k0 < K; k0 += GBK, ++itn) {
        const int cur = itn & 1;
        if (k0 + GBK < K) {
            if (tdmWave) tdm_load_tile(A + (long)rowBase * K + (k0 + GBK), &sA[cur ^ 1][0], K);
            stageB(cur ^ 1, k0 + GBK);          // prefetch next tile into other buffer
            if (tdmWave) __builtin_amdgcn_s_wait_tensorcnt(1);  // oldest TDM (cur A tile) done
            wait_async2();                       // oldest 2 asyncs (cur B tile) done
            if (k0 + 2 * GBK < K)                // keep L2 warm two tiles ahead
                __builtin_prefetch(Bt + (long)(colBase + (tid >> 2)) * K + k0 + 2 * GBK, 0, 0);
        } else {
            if (tdmWave) __builtin_amdgcn_s_wait_tensorcnt(0);
            wait_async0();
        }
        __syncthreads();                // current tile visible to all waves

        // A fragment: lanes 0-15 hold K {0..7,16..23}; lanes 16-31 {8..15,24..31}
        v16h aF[2];
        const int kOffA = halfSel * 8;
        #pragma unroll
        for (int mi = 0; mi < 2; ++mi) {
            const _Float16* src = &sA[cur][(wRow + mi * 16 + mLane) * LP];
            *reinterpret_cast<v8h*>(&aF[mi])       = *reinterpret_cast<const v8h*>(src + kOffA);
            *(reinterpret_cast<v8h*>(&aF[mi]) + 1) = *reinterpret_cast<const v8h*>(src + kOffA + 16);
        }
        // B fragment: lanes 0-15 hold K {0..7,16..23}; lanes 16-31 {8..15,24..31}
        v16h bF[4];
        const int kOffB = halfSel * 8;
        #pragma unroll
        for (int ni = 0; ni < 4; ++ni) {
            const _Float16* src = &sB[cur][(wCol + ni * 16 + mLane) * LP];
            *reinterpret_cast<v8h*>(&bF[ni])       = *reinterpret_cast<const v8h*>(src + kOffB);
            *(reinterpret_cast<v8h*>(&bF[ni]) + 1) = *reinterpret_cast<const v8h*>(src + kOffB + 16);
        }
        #pragma unroll
        for (int mi = 0; mi < 2; ++mi)
            #pragma unroll
            for (int ni = 0; ni < 4; ++ni)
                acc[mi][ni] = __builtin_amdgcn_wmma_f32_16x16x32_f16(
                    false, aF[mi], false, bF[ni], (short)0, acc[mi][ni], false, false);

        __syncthreads();                // all waves done reading before buffer reuse
    }

    // epilogue: C/D layout — VGPR r: lanes0-15 M=r, lanes16-31 M=r+8; N = lane%16
    #pragma unroll
    for (int ni = 0; ni < 4; ++ni) {
        int col = colBase + wCol + ni * 16 + mLane;
        float bv = bias ? bias[col] : 0.0f;
        #pragma unroll
        for (int mi = 0; mi < 2; ++mi) {
            int rb = rowBase + wRow + mi * 16 + halfSel * 8;
            #pragma unroll
            for (int r = 0; r < 8; ++r) {
                float v = acc[mi][ni][r] + bv;
                if (RELU) v = fmaxf(v, 0.0f);
                C[(long)(rb + r) * Nn + col] = v;
            }
        }
    }
}

// ---------------------------------------------------------------- LayerNorm + ReLU -> f16
__global__ __launch_bounds__(256)
void layernorm_relu_kernel(const float* __restrict__ X, const float* __restrict__ g,
                           const float* __restrict__ b, _Float16* __restrict__ out, int Hn) {
    int row = blockIdx.x;
    const float* x = X + (long)row * Hn;
    float s = 0.0f, s2 = 0.0f;
    for (int i = threadIdx.x; i < Hn; i += 256) { float v = x[i]; s += v; s2 += v * v; }
    for (int off = 16; off > 0; off >>= 1) { s += __shfl_down(s, off); s2 += __shfl_down(s2, off); }
    __shared__ float w1s[8], w2s[8];
    if ((threadIdx.x & 31) == 0) { w1s[threadIdx.x >> 5] = s; w2s[threadIdx.x >> 5] = s2; }
    __syncthreads();
    if (threadIdx.x == 0) {
        float t = 0.0f, t2 = 0.0f;
        for (int i = 0; i < 8; ++i) { t += w1s[i]; t2 += w2s[i]; }
        w1s[0] = t; w2s[0] = t2;
    }
    __syncthreads();
    float mu  = w1s[0] / Hn;
    float var = w2s[0] / Hn - mu * mu;
    float inv = rsqrtf(var + 1e-5f);
    for (int i = threadIdx.x; i < Hn; i += 256) {
        float v = (x[i] - mu) * inv * g[i] + b[i];
        out[(long)row * Hn + i] = (_Float16)fmaxf(v, 0.0f);
    }
}

// dcoef = sigmoid(dfraw[row,:] . w2 + b2), dfraw already relu'd
__global__ __launch_bounds__(256)
void dcoef_kernel(const float* __restrict__ dfraw, const float* __restrict__ w2v,
                  const float* __restrict__ b2, float* __restrict__ dcoef) {
    int row = blockIdx.x;
    float s = dfraw[(long)row * DF + threadIdx.x] * w2v[threadIdx.x];
    for (int off = 16; off > 0; off >>= 1) s += __shfl_down(s, off);
    __shared__ float w[8];
    if ((threadIdx.x & 31) == 0) w[threadIdx.x >> 5] = s;
    __syncthreads();
    if (threadIdx.x == 0) {
        float t = 0.0f;
        for (int i = 0; i < 8; ++i) t += w[i];
        dcoef[row] = 1.0f / (1.0f + expf(-(t + b2[0])));
    }
}

// k = dhdt - dcoef * (y - adjS)   (since L*y = y - adj*y)
__global__ void combine_k_kernel(const float* __restrict__ dhdt, const float* __restrict__ dcoef,
                                 const float* __restrict__ y, const float* __restrict__ adjS,
                                 float* __restrict__ kout, long n) {
    long i = (long)blockIdx.x * blockDim.x + threadIdx.x;
    if (i >= n) return;
    long row = i / DF;
    kout[i] = dhdt[i] - dcoef[row] * (y[i] - adjS[i]);
}

__global__ void axpy5_kernel(const float* __restrict__ y,
                             const float* __restrict__ k0, const float* __restrict__ k1,
                             const float* __restrict__ k2, const float* __restrict__ k3,
                             const float* __restrict__ k4,
                             float c0, float c1, float c2, float c3, float c4,
                             float* __restrict__ out, long n) {
    long i = (long)blockIdx.x * blockDim.x + threadIdx.x;
    if (i >= n) return;
    out[i] = y[i] + c0 * k0[i] + c1 * k1[i] + c2 * k2[i] + c3 * k3[i] + c4 * k4[i];
}

// ---------------------------------------------------------------- host side
namespace {
struct Ctx {
    const float *state, *context, *edge_attr;
    const int*   edge_index;
    const float *ew_w1, *ew_b1, *ew_w2, *ew_b2;
    const float *td_w1, *td_b1, *ln1_g, *ln1_b, *td_w2, *td_b2, *ln2_g, *ln2_b, *td_w3, *td_b3;
    const float *df_w1, *df_b1, *df_w2, *df_b2;
    _Float16 *adjh, *w1T, *w2T, *w3T, *dfw1T, *Xh, *Hact, *yh, *yhT;
    float *adjf, *ew, *deg, *dis, *G1, *dhdt, *dfraw, *dcoef, *adjS, *ycur, *yib, *ks;
    hipStream_t stream;
};

inline long cdiv(long a, long b) { return (a + b - 1) / b; }

template <bool RELU>
void launch_gemm(const Ctx& c, const _Float16* A, const _Float16* Bt, const float* bias,
                 float* C, int M, int N, int K, int Z, long sAz, long sBz, long sCz) {
    dim3 g((unsigned)(N / GBN), (unsigned)(M / GBM), (unsigned)Z);
    wmma_gemm_kernel<RELU><<<g, 256, 0, c.stream>>>(A, Bt, bias, C, M, N, K, sAz, sBz, sCz);
}

void feval(const Ctx& c, const float* yi, float* kout) {
    const long nyf = (long)DR * DF;
    const long nxf = (long)DR * 2 * DF;
    cvt_f32_f16_kernel<<<cdiv(nyf, 256), 256, 0, c.stream>>>(yi, c.yh, nyf);
    transpose_cvt_kernel<<<dim3(DF / 32, DN / 32, DB), 256, 0, c.stream>>>(
        yi, c.yhT, DN, DF, (long)DN * DF, (long)DF * DN);
    build_x_kernel<<<cdiv(nxf, 256), 256, 0, c.stream>>>(yi, c.context, c.Xh);
    // temporal MLP
    launch_gemm<false>(c, c.Xh, c.w1T, c.td_b1, c.G1, DR, DH, 2 * DF, 1, 0, 0, 0);
    layernorm_relu_kernel<<<DR, 256, 0, c.stream>>>(c.G1, c.ln1_g, c.ln1_b, c.Hact, DH);
    launch_gemm<false>(c, c.Hact, c.w2T, c.td_b2, c.G1, DR, DH, DH, 1, 0, 0, 0);
    layernorm_relu_kernel<<<DR, 256, 0, c.stream>>>(c.G1, c.ln2_g, c.ln2_b, c.Hact, DH);
    launch_gemm<false>(c, c.Hact, c.w3T, c.td_b3, c.dhdt, DR, DF, DH, 1, 0, 0, 0);
    // diffusion coefficient branch
    launch_gemm<true>(c, c.yh, c.dfw1T, c.df_b1, c.dfraw, DR, DF, DF, 1, 0, 0, 0);
    dcoef_kernel<<<DR, 256, 0, c.stream>>>(c.dfraw, c.df_w2, c.df_b2, c.dcoef);
    // batched adj @ state
    launch_gemm<false>(c, c.adjh, c.yhT, nullptr, c.adjS, DN, DF, DN, DB,
                       0, (long)DF * DN, (long)DN * DF);
    combine_k_kernel<<<cdiv(nyf, 256), 256, 0, c.stream>>>(c.dhdt, c.dcoef, yi, c.adjS, kout, nyf);
}
} // namespace

extern "C" void kernel_launch(void* const* d_in, const int* in_sizes, int n_in,
                              void* d_out, int out_size, void* d_ws, size_t ws_size,
                              hipStream_t stream) {
    Ctx c{};
    c.stream     = stream;
    c.state      = (const float*)d_in[0];
    c.context    = (const float*)d_in[1];
    c.edge_attr  = (const float*)d_in[2];
    c.edge_index = (const int*)  d_in[3];
    c.ew_w1 = (const float*)d_in[4];  c.ew_b1 = (const float*)d_in[5];
    c.ew_w2 = (const float*)d_in[6];  c.ew_b2 = (const float*)d_in[7];
    c.td_w1 = (const float*)d_in[8];  c.td_b1 = (const float*)d_in[9];
    c.ln1_g = (const float*)d_in[10]; c.ln1_b = (const float*)d_in[11];
    c.td_w2 = (const float*)d_in[12]; c.td_b2 = (const float*)d_in[13];
    c.ln2_g = (const float*)d_in[14]; c.ln2_b = (const float*)d_in[15];
    c.td_w3 = (const float*)d_in[16]; c.td_b3 = (const float*)d_in[17];
    c.df_w1 = (const float*)d_in[18]; c.df_b1 = (const float*)d_in[19];
    c.df_w2 = (const float*)d_in[20]; c.df_b2 = (const float*)d_in[21];

    char* p = (char*)d_ws;
    auto alloc = [&](size_t bytes) -> void* {
        void* r = (void*)p;
        p += (bytes + 255) & ~(size_t)255;
        return r;
    };
    const long RF = (long)DR * DF;
    c.G1    = (float*)alloc((size_t)DR * DH * 4);        // reused as adjf scratch
    c.adjf  = c.G1;
    c.adjh  = (_Float16*)alloc((size_t)DN * DN * 2);
    c.ew    = (float*)alloc((size_t)DE * 4);
    c.deg   = (float*)alloc((size_t)DN * 4);
    c.dis   = (float*)alloc((size_t)DN * 4);
    c.w1T   = (_Float16*)alloc((size_t)2 * DF * DH * 2);
    c.w2T   = (_Float16*)alloc((size_t)DH * DH * 2);
    c.w3T   = (_Float16*)alloc((size_t)DH * DF * 2);
    c.dfw1T = (_Float16*)alloc((size_t)DF * DF * 2);
    c.Xh    = (_Float16*)alloc((size_t)DR * 2 * DF * 2);
    c.Hact  = (_Float16*)alloc((size_t)DR * DH * 2);
    c.yh    = (_Float16*)alloc((size_t)RF * 2);
    c.yhT   = (_Float16*)alloc((size_t)RF * 2);
    c.dhdt  = (float*)alloc((size_t)RF * 4);
    c.dfraw = (float*)alloc((size_t)RF * 4);
    c.dcoef = (float*)alloc((size_t)DR * 4);
    c.adjS  = (float*)alloc((size_t)RF * 4);
    c.ycur  = (float*)alloc((size_t)RF * 4);
    c.yib   = (float*)alloc((size_t)RF * 4);
    c.ks    = (float*)alloc((size_t)6 * RF * 4);
    (void)ws_size; (void)n_in; (void)in_sizes; (void)out_size;

    // ---- one-time per call: transposed f16 weights, adjacency build ---------
    transpose_cvt_kernel<<<dim3(DH / 32, (2 * DF) / 32, 1), 256, 0, stream>>>(
        c.td_w1, c.w1T, 2 * DF, DH, 0, 0);
    transpose_cvt_kernel<<<dim3(DH / 32, DH / 32, 1), 256, 0, stream>>>(
        c.td_w2, c.w2T, DH, DH, 0, 0);
    transpose_cvt_kernel<<<dim3(DF / 32, DH / 32, 1), 256, 0, stream>>>(
        c.td_w3, c.w3T, DH, DF, 0, 0);
    transpose_cvt_kernel<<<dim3(DF / 32, DF / 32, 1), 256, 0, stream>>>(
        c.df_w1, c.dfw1T, DF, DF, 0, 0);

    edge_mlp_kernel<<<DE / 256, 256, 0, stream>>>(c.edge_attr, c.ew_w1, c.ew_b1, c.ew_w2, c.ew_b2, c.ew);
    fill_f32_kernel<<<cdiv(DN, 256), 256, 0, stream>>>(c.deg, 1.0f, DN);
    deg_acc_kernel<<<DE / 256, 256, 0, stream>>>(c.edge_index, c.deg);
    fill_f32_kernel<<<cdiv((long)DN * DN, 256), 256, 0, stream>>>(c.adjf, 0.0f, (long)DN * DN);
    dis_diag_kernel<<<cdiv(DN, 256), 256, 0, stream>>>(c.deg, c.dis, c.adjf);
    adj_acc_kernel<<<DE / 256, 256, 0, stream>>>(c.edge_index, c.ew, c.dis, c.adjf);
    cvt_f32_f16_kernel<<<cdiv((long)DN * DN, 256), 256, 0, stream>>>(c.adjf, c.adjh, (long)DN * DN);

    copy_f32_kernel<<<cdiv(RF, 256), 256, 0, stream>>>(c.state, c.ycur, RF);

    // ---- dopri5 fixed-step (k7 dropped: b7 == 0) ----------------------------
    const float dt = (1.0f - 0.0f) / 4.0f;
    float* K0 = c.ks + 0 * RF; float* K1 = c.ks + 1 * RF; float* K2 = c.ks + 2 * RF;
    float* K3 = c.ks + 3 * RF; float* K4 = c.ks + 4 * RF; float* K5 = c.ks + 5 * RF;
    const long nb = cdiv(RF, 256);

    for (int step = 0; step < 4; ++step) {
        feval(c, c.ycur, K0);
        axpy5_kernel<<<nb, 256, 0, stream>>>(c.ycur, K0, K0, K0, K0, K0,
            dt * (1.0f/5.0f), 0.f, 0.f, 0.f, 0.f, c.yib, RF);
        feval(c, c.yib, K1);
        axpy5_kernel<<<nb, 256, 0, stream>>>(c.ycur, K0, K1, K0, K0, K0,
            dt * (3.0f/40.0f), dt * (9.0f/40.0f), 0.f, 0.f, 0.f, c.yib, RF);
        feval(c, c.yib, K2);
        axpy5_kernel<<<nb, 256, 0, stream>>>(c.ycur, K0, K1, K2, K0, K0,
            dt * (44.0f/45.0f), dt * (-56.0f/15.0f), dt * (32.0f/9.0f), 0.f, 0.f, c.yib, RF);
        feval(c, c.yib, K3);
        axpy5_kernel<<<nb, 256, 0, stream>>>(c.ycur, K0, K1, K2, K3, K0,
            dt * (19372.0f/6561.0f), dt * (-25360.0f/2187.0f),
            dt * (64448.0f/6561.0f), dt * (-212.0f/729.0f), 0.f, c.yib, RF);
        feval(c, c.yib, K4);
        axpy5_kernel<<<nb, 256, 0, stream>>>(c.ycur, K0, K1, K2, K3, K4,
            dt * (9017.0f/3168.0f), dt * (-355.0f/33.0f), dt * (46732.0f/5247.0f),
            dt * (49.0f/176.0f), dt * (-5103.0f/18656.0f), c.yib, RF);
        feval(c, c.yib, K5);
        axpy5_kernel<<<nb, 256, 0, stream>>>(c.ycur, K0, K2, K3, K4, K5,
            dt * (35.0f/384.0f), dt * (500.0f/1113.0f), dt * (125.0f/192.0f),
            dt * (-2187.0f/6784.0f), dt * (11.0f/84.0f), c.ycur, RF);
    }

    copy_f32_kernel<<<cdiv(RF, 256), 256, 0, stream>>>(c.ycur, (float*)d_out, RF);
}
